// SelectiveSSM_42322607735162
// MI455X (gfx1250) — matrix operand order repaired and checked by reading, not compiled, not run on hardware
//
#include <hip/hip_runtime.h>

// Problem sizes (fixed by the reference)
#define B_SZ   2
#define LSEQ   2048
#define DMODEL 1024
#define NST    16
// Chunked-scan configuration
#define CHUNKS 32
#define CLEN   64     // LSEQ / CHUNKS
#define TS     16     // timesteps staged in LDS per barrier

typedef __attribute__((ext_vector_type(16))) __bf16 v16bf;
typedef __attribute__((ext_vector_type(8)))  float  v8f;

union FragBF {
    v16bf v;
    uint4 q[2];
};

__device__ __forceinline__ unsigned short f32_to_bf16_rne(float f) {
    union { float f; unsigned int u; } c; c.f = f;
    unsigned int u = c.u;
    unsigned int r = u + 0x7FFFu + ((u >> 16) & 1u);   // round to nearest even
    return (unsigned short)(r >> 16);
}

// ---------------- precompute kernels ----------------

__global__ void k_cvt_bf16(const float* __restrict__ src,
                           unsigned short* __restrict__ dst, int n) {
    int i = blockIdx.x * blockDim.x + threadIdx.x;
    if (i < n) dst[i] = f32_to_bf16_rne(src[i]);
}

__global__ void k_atab(const float* __restrict__ logA,
                       float* __restrict__ Atab, int n) {
    int i = blockIdx.x * blockDim.x + threadIdx.x;
    if (i < n) Atab[i] = -__expf(logA[i]);     // A = -exp(log_A)
}

// ---------------- WMMA projection GEMM ----------------
// out rows M = B*L = 4096. Per wave:
//   ng in [0,16): 16x64 tile of delta_pre = Xbf * Wd^T, fused softplus+bias
//   ng == 16   : 16x16 Bm tile and 16x16 Cm tile (A-fragment reused)
__global__ __launch_bounds__(128)
void k_gemm_wmma(const unsigned short* __restrict__ Xbf,
                 const unsigned short* __restrict__ Wd,
                 const unsigned short* __restrict__ Wb,
                 const unsigned short* __restrict__ Wc,
                 const float* __restrict__ bdelta,
                 float* __restrict__ delta,
                 float* __restrict__ Bm,
                 float* __restrict__ Cm) {
    const int NGROUPS = DMODEL / 64 + 1;                       // 17
    int wave = blockIdx.x * (blockDim.x >> 5) + (threadIdx.x >> 5);
    int lane = threadIdx.x & 31;
    int mt = wave / NGROUPS;
    int ng = wave % NGROUPS;
    int hi = (lane >> 4) & 1;                                  // lane >= 16 ?
    int row = mt * 16 + (lane & 15);
    const uint4* arow = (const uint4*)(Xbf + (size_t)row * DMODEL);

    if (ng < 16) {
        v8f acc[4] = {};
        for (int k0 = 0; k0 < DMODEL; k0 += 32) {
            // A fragment: 16x32 bf16, contiguous k = 16*hi + s (B uses the same order)
            FragBF a;
            int au = (k0 >> 3) + 2 * hi;   // uint4 = 8 bf16
            a.q[0] = arow[au];
            a.q[1] = arow[au + 1];
            int bu = (k0 >> 3) + 2 * hi;
#pragma unroll
            for (int j = 0; j < 4; ++j) {
                int col = (ng * 4 + j) * 16 + (lane & 15);
                const uint4* brow = (const uint4*)(Wd + (size_t)col * DMODEL);
                FragBF b;
                b.q[0] = brow[bu];
                b.q[1] = brow[bu + 1];
                acc[j] = __builtin_amdgcn_wmma_f32_16x16x32_bf16(
                    false, a.v, false, b.v, (short)0, acc[j], false, false);
            }
        }
        // fused bias + softplus epilogue
#pragma unroll
        for (int j = 0; j < 4; ++j) {
            int col = (ng * 4 + j) * 16 + (lane & 15);
            float bd = bdelta[col];
#pragma unroll
            for (int i = 0; i < 8; ++i) {
                int r = mt * 16 + i + 8 * hi;
                float z = acc[j][i] + bd;
                float sp = (z > 20.f) ? z : log1pf(__expf(z));
                delta[(size_t)r * DMODEL + col] = sp;
            }
        }
    } else {
        // Bm and Cm: N = 16, share the A fragment
        v8f accB = {}, accC = {};
        int col = lane & 15;
        const uint4* bbrow = (const uint4*)(Wb + (size_t)col * DMODEL);
        const uint4* bcrow = (const uint4*)(Wc + (size_t)col * DMODEL);
        for (int k0 = 0; k0 < DMODEL; k0 += 32) {
            FragBF a;
            int au = (k0 >> 3) + 2 * hi;
            a.q[0] = arow[au];
            a.q[1] = arow[au + 1];
            int bu = (k0 >> 3) + 2 * hi;
            FragBF fb, fc;
            fb.q[0] = bbrow[bu]; fb.q[1] = bbrow[bu + 1];
            fc.q[0] = bcrow[bu]; fc.q[1] = bcrow[bu + 1];
            accB = __builtin_amdgcn_wmma_f32_16x16x32_bf16(
                false, a.v, false, fb.v, (short)0, accB, false, false);
            accC = __builtin_amdgcn_wmma_f32_16x16x32_bf16(
                false, a.v, false, fc.v, (short)0, accC, false, false);
        }
#pragma unroll
        for (int i = 0; i < 8; ++i) {
            int r = mt * 16 + i + 8 * hi;
            Bm[(size_t)r * NST + col] = accB[i];
            Cm[(size_t)r * NST + col] = accC[i];
        }
    }
}

// ---------------- chunked linear scan ----------------
// Pass 1: per chunk, per (b,d) channel, compute P[n] = prod(a_t) and
// local[n] = scan starting from h=0.
__global__ __launch_bounds__(256)
void k_scan1(const float* __restrict__ delta, const float* __restrict__ x,
             const float* __restrict__ Bm, const float* __restrict__ Atab,
             float* __restrict__ CP, float* __restrict__ CL) {
    __shared__ float sB[TS * NST];
    int tid  = threadIdx.x;
    int blk  = blockIdx.x;             // ((b*CHUNKS + chunk)*4 + dblk)
    int dblk = blk & 3;
    int bc   = blk >> 2;
    int chunk = bc % CHUNKS;
    int b     = bc / CHUNKS;
    int d     = dblk * 256 + tid;

    float Arow[NST], P[NST], Lc[NST];
#pragma unroll
    for (int n = 0; n < NST; ++n) {
        Arow[n] = Atab[d * NST + n];
        P[n] = 1.f; Lc[n] = 0.f;
    }

    int t0base = chunk * CLEN;
    for (int s = 0; s < CLEN; s += TS) {
        int t0 = t0base + s;
        __syncthreads();
        {   // stage Bm rows t0..t0+TS-1 (TS*NST == blockDim)
            int tt = tid >> 4, nn = tid & 15;
            sB[tid] = Bm[(size_t)(b * LSEQ + t0 + tt) * NST + nn];
        }
        __syncthreads();
        for (int t = 0; t < TS; ++t) {
            size_t gi = (size_t)(b * LSEQ + t0 + t) * DMODEL + d;
            float dv = delta[gi];
            float xv = x[gi];
            float dx = dv * xv;
#pragma unroll
            for (int n = 0; n < NST; ++n) {
                float a = __expf(dv * Arow[n]);
                P[n] *= a;
                Lc[n] = a * Lc[n] + dx * sB[t * NST + n];
            }
        }
    }
    size_t o = (size_t)((b * CHUNKS + chunk) * DMODEL + d) * NST;
#pragma unroll
    for (int n = 0; n < NST; ++n) { CP[o + n] = P[n]; CL[o + n] = Lc[n]; }
}

// Middle: sequential combine over 32 chunks per (b,d,n); write exclusive
// start state for each chunk.
__global__ __launch_bounds__(256)
void k_mid(const float* __restrict__ CP, const float* __restrict__ CL,
           float* __restrict__ Hs) {
    int t = blockIdx.x * blockDim.x + threadIdx.x;      // B*DM*N = 32768
    int b = t / (DMODEL * NST);
    int r = t % (DMODEL * NST);
    float h = 0.f;
    for (int c = 0; c < CHUNKS; ++c) {
        size_t idx = (size_t)(b * CHUNKS + c) * DMODEL * NST + r;
        Hs[idx] = h;
        h = CP[idx] * h + CL[idx];
    }
}

// Pass 2: replay each chunk from its corrected start state, emit y.
__global__ __launch_bounds__(256)
void k_scan2(const float* __restrict__ delta, const float* __restrict__ x,
             const float* __restrict__ Bm, const float* __restrict__ Cm,
             const float* __restrict__ Atab, const float* __restrict__ Hs,
             const float* __restrict__ Dv, float* __restrict__ out) {
    __shared__ float sB[TS * NST];
    __shared__ float sC[TS * NST];
    int tid  = threadIdx.x;
    int blk  = blockIdx.x;
    int dblk = blk & 3;
    int bc   = blk >> 2;
    int chunk = bc % CHUNKS;
    int b     = bc / CHUNKS;
    int d     = dblk * 256 + tid;

    float Arow[NST], h[NST];
    size_t ho = (size_t)((b * CHUNKS + chunk) * DMODEL + d) * NST;
#pragma unroll
    for (int n = 0; n < NST; ++n) {
        Arow[n] = Atab[d * NST + n];
        h[n] = Hs[ho + n];
    }
    float Dd = Dv[d];

    int t0base = chunk * CLEN;
    for (int s = 0; s < CLEN; s += TS) {
        int t0 = t0base + s;
        __syncthreads();
        {
            int tt = tid >> 4, nn = tid & 15;
            size_t ri = (size_t)(b * LSEQ + t0 + tt) * NST + nn;
            sB[tid] = Bm[ri];
            sC[tid] = Cm[ri];
        }
        __syncthreads();
        for (int t = 0; t < TS; ++t) {
            size_t gi = (size_t)(b * LSEQ + t0 + t) * DMODEL + d;
            float dv = delta[gi];
            float xv = x[gi];
            float dx = dv * xv;
            float y = 0.f;
#pragma unroll
            for (int n = 0; n < NST; ++n) {
                float a = __expf(dv * Arow[n]);
                h[n] = a * h[n] + dx * sB[t * NST + n];
                y += h[n] * sC[t * NST + n];
            }
            out[gi] = y + xv * Dd;
        }
    }
}

// ---------------- launch ----------------
extern "C" void kernel_launch(void* const* d_in, const int* in_sizes, int n_in,
                              void* d_out, int out_size, void* d_ws, size_t ws_size,
                              hipStream_t stream) {
    const float* x      = (const float*)d_in[0];   // (B,L,DM)
    const float* Wb     = (const float*)d_in[1];   // (N,DM)
    const float* Wc     = (const float*)d_in[2];   // (N,DM)
    const float* Wdelta = (const float*)d_in[3];   // (DM,DM)
    const float* bdelta = (const float*)d_in[4];   // (DM,)
    const float* logA   = (const float*)d_in[5];   // (DM,N)
    const float* Dvec   = (const float*)d_in[6];   // (DM,)
    float* out = (float*)d_out;

    char* w = (char*)d_ws;
    size_t off = 0;
    unsigned short* Xbf  = (unsigned short*)(w + off); off += (size_t)B_SZ * LSEQ * DMODEL * 2;
    unsigned short* Wdbf = (unsigned short*)(w + off); off += (size_t)DMODEL * DMODEL * 2;
    unsigned short* Wbbf = (unsigned short*)(w + off); off += (size_t)NST * DMODEL * 2;
    unsigned short* Wcbf = (unsigned short*)(w + off); off += (size_t)NST * DMODEL * 2;
    float* delta = (float*)(w + off); off += (size_t)B_SZ * LSEQ * DMODEL * 4;
    float* Bm    = (float*)(w + off); off += (size_t)B_SZ * LSEQ * NST * 4;
    float* Cm    = (float*)(w + off); off += (size_t)B_SZ * LSEQ * NST * 4;
    float* Atab  = (float*)(w + off); off += (size_t)DMODEL * NST * 4;
    float* CP    = (float*)(w + off); off += (size_t)B_SZ * CHUNKS * DMODEL * NST * 4;
    float* CL    = (float*)(w + off); off += (size_t)B_SZ * CHUNKS * DMODEL * NST * 4;
    float* Hs    = (float*)(w + off); off += (size_t)B_SZ * CHUNKS * DMODEL * NST * 4;

    // 1) precision conversion + A table
    int nx = B_SZ * LSEQ * DMODEL;
    k_cvt_bf16<<<(nx + 255) / 256, 256, 0, stream>>>(x, Xbf, nx);
    k_cvt_bf16<<<(DMODEL * DMODEL + 255) / 256, 256, 0, stream>>>(Wdelta, Wdbf, DMODEL * DMODEL);
    k_cvt_bf16<<<(NST * DMODEL + 255) / 256, 256, 0, stream>>>(Wb, Wbbf, NST * DMODEL);
    k_cvt_bf16<<<(NST * DMODEL + 255) / 256, 256, 0, stream>>>(Wc, Wcbf, NST * DMODEL);
    k_atab<<<(DMODEL * NST + 255) / 256, 256, 0, stream>>>(logA, Atab, DMODEL * NST);

    // 2) WMMA projections (delta with fused softplus, Bm, Cm)
    int jobs = (B_SZ * LSEQ / 16) * (DMODEL / 64 + 1);   // 256 * 17 = 4352 waves
    k_gemm_wmma<<<jobs / 4, 128, 0, stream>>>(Xbf, Wdbf, Wbbf, Wcbf, bdelta,
                                              delta, Bm, Cm);

    // 3) chunked linear scan (32-way parallel over L)
    k_scan1<<<B_SZ * CHUNKS * 4, 256, 0, stream>>>(delta, x, Bm, Atab, CP, CL);
    k_mid<<<(B_SZ * DMODEL * NST) / 256, 256, 0, stream>>>(CP, CL, Hs);
    k_scan2<<<B_SZ * CHUNKS * 4, 256, 0, stream>>>(delta, x, Bm, Cm, Atab, Hs,
                                                   Dvec, out);
}
